// our_adaptive_45775761441079
// MI455X (gfx1250) — matrix operand order repaired and checked by reading, not compiled, not run on hardware
//
#include <hip/hip_runtime.h>
#include <math.h>
#include <stdint.h>

typedef __attribute__((ext_vector_type(16))) _Float16 v16h;
typedef __attribute__((ext_vector_type(8)))  float    v8f;
typedef __attribute__((ext_vector_type(4)))  unsigned int v4u;
typedef __attribute__((ext_vector_type(8)))  int      v8i;
typedef __attribute__((ext_vector_type(4)))  int      v4i;

union HFrag { v16h v; uint4 q[2]; _Float16 h[16]; };

__device__ __forceinline__ float sigmf(float x) { return 1.0f / (1.0f + expf(-x)); }

// ------------------------------------------------------------------
// Elementwise utility kernels
// ------------------------------------------------------------------
__global__ void k_cvt_f16(_Float16* __restrict__ dst, const float* __restrict__ src,
                          int n, int absv) {
  int i = blockIdx.x * blockDim.x + threadIdx.x;
  if (i < n) { float v = src[i]; if (absv) v = fabsf(v); dst[i] = (_Float16)v; }
}
__global__ void k_zero(float* __restrict__ p, int n) {
  int i = blockIdx.x * blockDim.x + threadIdx.x;
  if (i < n) p[i] = 0.0f;
}
__global__ void k_copy2(float* __restrict__ d1, float* __restrict__ d2,
                        const float* __restrict__ s, int n) {
  int i = blockIdx.x * blockDim.x + threadIdx.x;
  if (i < n) { float v = s[i]; d1[i] = v; d2[i] = v; }
}
__global__ void k_add(float* __restrict__ dst, const float* __restrict__ src, int n) {
  int i = blockIdx.x * blockDim.x + threadIdx.x;
  if (i < n) dst[i] += src[i];
}

// ------------------------------------------------------------------
// Edge-gate MLP: per 16-edge tile, [16,256]x[256,128] WMMA GEMM,
// ReLU, dot with w2, logistic-noise gate for BOTH value streams.
// W1 (64KB f16) is staged to LDS once per block via the Tensor Data
// Mover (tensor_load_to_lds + s_wait_tensorcnt); A-fragments are
// gathered straight from global (f32) and converted in registers.
// blockDim = 128 (4 waves), one 16-edge tile per wave.
// ------------------------------------------------------------------
#define GW 4
__global__ __launch_bounds__(128) void k_gate(
    const int* __restrict__ eu, const int* __restrict__ ei,
    const float* __restrict__ val_ui, const float* __restrict__ val_iu,
    const float* __restrict__ eps_ui, const float* __restrict__ eps_iu,
    const float* __restrict__ uemb, const float* __restrict__ iemb,
    const _Float16* __restrict__ W1h, const float* __restrict__ b1,
    const float* __restrict__ W2, const float* __restrict__ b2,
    float* __restrict__ v_ui, float* __restrict__ v_iu, int ne)
{
  __shared__ __align__(128) _Float16 sW1[128 * 256];   // 64 KB: whole W1, TDM-loaded

  // ---- TDM: one descriptor, whole 128x256 f16 tile -> LDS ----
  if (threadIdx.x < 32) {
    unsigned lds = (unsigned)(uintptr_t)(&sW1[0]);     // LDS aperture: low 32 bits
    unsigned long long ga = (unsigned long long)(uintptr_t)W1h;
    v4u g0 = { 1u,                                     // count=1 user descriptor
               lds,                                    // lds_addr (bytes)
               (unsigned)(ga & 0xFFFFFFFFu),           // global_addr[31:0]
               (unsigned)((ga >> 32) & 0x01FFFFFFu) | (2u << 30) };  // addr[56:32], type=2
    v8i g1 = { (int)(1u << 16),                        // data_size = 2 bytes
               (int)(256u << 16),                      // tensor_dim0 = 256 (lo16 @ 63:48)
               (int)(128u << 16),                      // dim0 hi=0 | tensor_dim1=128 (lo16)
               (int)(256u << 16),                      // dim1 hi=0 | tile_dim0 = 256
               (int)128,                               // tile_dim1 = 128, tile_dim2 = 0
               (int)256,                               // tensor_dim0_stride = 256
               0, 0 };
    v4i gz = { 0, 0, 0, 0 };
#if __has_include(<hip/amd_detail/amd_gfx1250_TDM.h>)
    v8i gz8 = { 0, 0, 0, 0, 0, 0, 0, 0 };
    __builtin_amdgcn_tensor_load_to_lds(g0, g1, gz, gz, gz8, 0);
#else
    __builtin_amdgcn_tensor_load_to_lds(g0, g1, gz, gz, 0);
#endif
    __builtin_amdgcn_s_wait_tensorcnt((short)0);
  }
  __syncthreads();

  const int wid  = threadIdx.x >> 5;
  const int lane = threadIdx.x & 31;
  const int base = (blockIdx.x * GW + wid) * 16;       // ne % (16*GW) == 0
  const int m  = lane & 15;
  const int hi = lane >> 4;

  const int e0 = base + m;
  const float* uRow = uemb + (size_t)eu[e0] * 128;
  const float* iRow = iemb + (size_t)ei[e0] * 128;

  v8f zero8 = {0.f, 0.f, 0.f, 0.f, 0.f, 0.f, 0.f, 0.f};
  v8f acc[8];
#pragma unroll
  for (int i = 0; i < 8; ++i) acc[i] = zero8;

#pragma unroll
  for (int kt = 0; kt < 8; ++kt) {                     // K = 256 in chunks of 32
    const float* src = (kt < 4) ? (uRow + kt * 32) : (iRow + (kt - 4) * 32);
    float4 f0 = *(const float4*)(src + hi * 8);
    float4 f1 = *(const float4*)(src + hi * 8 + 4);
    float4 f2 = *(const float4*)(src + hi * 8 + 16);
    float4 f3 = *(const float4*)(src + hi * 8 + 20);
    HFrag a;
    a.h[0]  = (_Float16)f0.x; a.h[1]  = (_Float16)f0.y;
    a.h[2]  = (_Float16)f0.z; a.h[3]  = (_Float16)f0.w;
    a.h[4]  = (_Float16)f1.x; a.h[5]  = (_Float16)f1.y;
    a.h[6]  = (_Float16)f1.z; a.h[7]  = (_Float16)f1.w;
    a.h[8]  = (_Float16)f2.x; a.h[9]  = (_Float16)f2.y;
    a.h[10] = (_Float16)f2.z; a.h[11] = (_Float16)f2.w;
    a.h[12] = (_Float16)f3.x; a.h[13] = (_Float16)f3.y;
    a.h[14] = (_Float16)f3.z; a.h[15] = (_Float16)f3.w;
#pragma unroll
    for (int nt = 0; nt < 8; ++nt) {                   // N = 128 hidden in tiles of 16
      HFrag b;
      const _Float16* wrow = sW1 + (nt * 16 + m) * 256 + kt * 32 + hi * 8;
      b.q[0] = *(const uint4*)(wrow);
      b.q[1] = *(const uint4*)(wrow + 16);
      acc[nt] = __builtin_amdgcn_wmma_f32_16x16x32_f16(
          false, a.v, false, b.v, (short)0, acc[nt], false, false);
    }
  }

  // Epilogue: relu(h + b1) dot w2, reduced over the 16 lanes per half
  float p[8];
#pragma unroll
  for (int r = 0; r < 8; ++r) p[r] = 0.0f;
#pragma unroll
  for (int nt = 0; nt < 8; ++nt) {
    int n = nt * 16 + m;
    float bn = b1[n], wn = W2[n];
#pragma unroll
    for (int r = 0; r < 8; ++r) {
      float h = fmaxf(acc[nt][r] + bn, 0.0f);
      p[r] += h * wn;
    }
  }
#pragma unroll
  for (int r = 0; r < 8; ++r)
#pragma unroll
    for (int mask = 1; mask < 16; mask <<= 1)
      p[r] += __shfl_xor(p[r], mask, 16);

  float bb = b2[0];
  if (m < 8) {                                     // lanes 0-7 -> rows 0-7; 16-23 -> 8-15
    int e = base + m + 8 * hi;
    float logit = p[m] + bb;
    float e1 = fminf(fmaxf(eps_ui[e], 1e-6f), 1.0f - 1e-6f);
    float g1 = logf(e1) - log1pf(-e1);
    v_ui[e] = val_ui[e] * (0.3f * sigmf((logit + g1) * 5.0f) + 0.7f);   // 1/TEMP = 5
    float e2 = fminf(fmaxf(eps_iu[e], 1e-6f), 1.0f - 1e-6f);
    float g2 = logf(e2) - log1pf(-e2);
    v_iu[e] = val_iu[e] * (0.3f * sigmf((logit + g2) * 5.0f) + 0.7f);
  }
}

// ------------------------------------------------------------------
// SPMM scatter: out[rows[e]] += vals[e] * x[cols[e]]  (D=128)
// One wave per edge; each lane handles 4 floats (float4 load, 4 atomics).
// ------------------------------------------------------------------
__global__ __launch_bounds__(256) void k_spmm(
    const int* __restrict__ rows, const int* __restrict__ cols,
    const float* __restrict__ vals, const float* __restrict__ x,
    float* __restrict__ out, int ne)
{
  int wave = blockIdx.x * (blockDim.x >> 5) + (threadIdx.x >> 5);
  int lane = threadIdx.x & 31;
  if (wave >= ne) return;
  int r = rows[wave], c = cols[wave];
  float v = vals[wave];
  float4 xv = *(const float4*)(x + (size_t)c * 128 + lane * 4);
  float* o = out + (size_t)r * 128 + lane * 4;
  atomicAdd(o + 0, v * xv.x);
  atomicAdd(o + 1, v * xv.y);
  atomicAdd(o + 2, v * xv.z);
  atomicAdd(o + 3, v * xv.w);
}

// ------------------------------------------------------------------
// Head stage 1: stat/diff GEMMs against knowledge_emb^T, produce
// x = sigmoid(disc) * (stat - diff) * kn_in   as f16 [B,128].
// ------------------------------------------------------------------
#define HW 4
__global__ __launch_bounds__(128) void k_head1(
    const int* __restrict__ stu_id, const int* __restrict__ exer_id,
    const float* __restrict__ kn_in,
    const float* __restrict__ stuSum, const float* __restrict__ exeSum,
    const _Float16* __restrict__ knh,
    const float* __restrict__ stu_b, const float* __restrict__ exer_b,
    const float* __restrict__ disc, _Float16* __restrict__ xh, int nrows)
{
  __shared__ __align__(16) _Float16 feat[HW][2][16 * 128];   // 32 KB
  const int wid  = threadIdx.x >> 5;
  const int lane = threadIdx.x & 31;
  const int base = (blockIdx.x * HW + wid) * 16;

  for (int t = lane; t < 16 * 32; t += 32) {
    int row = t >> 5, c4 = (t & 31) << 2;
    int b = base + row;
    float4 s = *(const float4*)(stuSum + (size_t)stu_id[b] * 128 + c4);
    _Float16* d = &feat[wid][0][row * 128 + c4];
    d[0] = (_Float16)s.x; d[1] = (_Float16)s.y; d[2] = (_Float16)s.z; d[3] = (_Float16)s.w;
    float4 e = *(const float4*)(exeSum + (size_t)exer_id[b] * 128 + c4);
    _Float16* d2 = &feat[wid][1][row * 128 + c4];
    d2[0] = (_Float16)e.x; d2[1] = (_Float16)e.y; d2[2] = (_Float16)e.z; d2[3] = (_Float16)e.w;
  }
  __syncthreads();

  const int m  = lane & 15;
  const int hi = lane >> 4;

  v8f zero8 = {0.f, 0.f, 0.f, 0.f, 0.f, 0.f, 0.f, 0.f};
  v8f accS[8], accD[8];
#pragma unroll
  for (int i = 0; i < 8; ++i) { accS[i] = zero8; accD[i] = zero8; }

  const _Float16* srow = &feat[wid][0][m * 128];
  const _Float16* erow = &feat[wid][1][m * 128];
#pragma unroll
  for (int kt = 0; kt < 4; ++kt) {
    HFrag aS, aD;
    int kb = kt * 32 + hi * 8;
    aS.q[0] = *(const uint4*)(srow + kb);  aS.q[1] = *(const uint4*)(srow + kb + 16);
    aD.q[0] = *(const uint4*)(erow + kb);  aD.q[1] = *(const uint4*)(erow + kb + 16);
#pragma unroll
    for (int nt = 0; nt < 8; ++nt) {
      HFrag b;
      const _Float16* wrow = knh + (size_t)(nt * 16 + m) * 128 + kt * 32 + hi * 8;
      b.q[0] = *(const uint4*)(wrow);
      b.q[1] = *(const uint4*)(wrow + 16);
      accS[nt] = __builtin_amdgcn_wmma_f32_16x16x32_f16(
          false, aS.v, false, b.v, (short)0, accS[nt], false, false);
      accD[nt] = __builtin_amdgcn_wmma_f32_16x16x32_f16(
          false, aD.v, false, b.v, (short)0, accD[nt], false, false);
    }
  }

  float sb[8], eb[8], dc[8];
#pragma unroll
  for (int r = 0; r < 8; ++r) {
    int b = base + r + 8 * hi;
    sb[r] = stu_b[stu_id[b]];
    eb[r] = exer_b[exer_id[b]];
    dc[r] = sigmf(disc[exer_id[b]]);
  }
#pragma unroll
  for (int nt = 0; nt < 8; ++nt) {
    int n = nt * 16 + m;
#pragma unroll
    for (int r = 0; r < 8; ++r) {
      int b = base + r + 8 * hi;
      float st = sigmf(0.25f * accS[nt][r] + sb[r]);   // /(L+1) folded in
      float df = sigmf(0.25f * accD[nt][r] + eb[r]);
      float xv = dc[r] * (st - df) * kn_in[(size_t)b * 128 + n];
      xh[(size_t)b * 128 + n] = (_Float16)xv;
    }
  }
}

// ------------------------------------------------------------------
// Generic f16 GEMM + bias + sigmoid:  out = sigmoid(in @ Wh^T + bias)
// in: [nrows,Kin] f16 row-major; Wh: [Nout,Kin] f16 (|W| pre-applied)
// ------------------------------------------------------------------
__global__ __launch_bounds__(128) void k_gemm_sig(
    const _Float16* __restrict__ inH, const _Float16* __restrict__ Wh,
    const float* __restrict__ bias, _Float16* __restrict__ outH,
    int Kin, int Nout, int nrows)
{
  const int wid  = threadIdx.x >> 5;
  const int lane = threadIdx.x & 31;
  const int base = (blockIdx.x * 4 + wid) * 16;
  const int m  = lane & 15;
  const int hi = lane >> 4;
  const int kts = Kin >> 5;
  const int nts = Nout >> 4;

  const _Float16* arow = inH + (size_t)(base + m) * Kin;
  v8f zero8 = {0.f, 0.f, 0.f, 0.f, 0.f, 0.f, 0.f, 0.f};

  for (int nt = 0; nt < nts; ++nt) {
    v8f acc = zero8;
    for (int kt = 0; kt < kts; ++kt) {
      HFrag a, b;
      int kb = kt * 32 + hi * 8;
      a.q[0] = *(const uint4*)(arow + kb);
      a.q[1] = *(const uint4*)(arow + kb + 16);
      const _Float16* wrow = Wh + (size_t)(nt * 16 + m) * Kin + kt * 32 + hi * 8;
      b.q[0] = *(const uint4*)(wrow);
      b.q[1] = *(const uint4*)(wrow + 16);
      acc = __builtin_amdgcn_wmma_f32_16x16x32_f16(
          false, a.v, false, b.v, (short)0, acc, false, false);
    }
    int n = nt * 16 + m;
    float bn = bias[n];
#pragma unroll
    for (int r = 0; r < 8; ++r) {
      int row = base + r + 8 * hi;
      outH[(size_t)row * Nout + n] = (_Float16)sigmf(acc[r] + bn);
    }
  }
}

// ------------------------------------------------------------------
// Final 128-dim dot with |pw3| + sigmoid
// ------------------------------------------------------------------
__global__ void k_out(const _Float16* __restrict__ h2, const float* __restrict__ pw3,
                      const float* __restrict__ pb3, float* __restrict__ out, int n)
{
  int b = blockIdx.x * blockDim.x + threadIdx.x;
  if (b >= n) return;
  const _Float16* r = h2 + (size_t)b * 128;
  float s = 0.0f;
  for (int k = 0; k < 128; ++k) s += (float)r[k] * fabsf(pw3[k]);
  out[b] = sigmf(s + pb3[0]);
}

// ------------------------------------------------------------------
extern "C" void kernel_launch(void* const* d_in, const int* in_sizes, int n_in,
                              void* d_out, int out_size, void* d_ws, size_t ws_size,
                              hipStream_t stream) {
  const int S = 50000, E = 20000, D = 128, NE = 800000, B = 8192;

  const int*   stu_id   = (const int*)  d_in[0];
  const int*   exer_id  = (const int*)  d_in[1];
  const float* kn_emb   = (const float*)d_in[2];
  const int*   ui1_u    = (const int*)  d_in[3];
  const int*   ui1_i    = (const int*)  d_in[4];
  const float* ui1_val  = (const float*)d_in[5];
  const float* iu1_val  = (const float*)d_in[6];
  const float* eps_ui1  = (const float*)d_in[7];
  const float* eps_iu1  = (const float*)d_in[8];
  const int*   ui0_u    = (const int*)  d_in[9];
  const int*   ui0_i    = (const int*)  d_in[10];
  const float* ui0_val  = (const float*)d_in[11];
  const float* iu0_val  = (const float*)d_in[12];
  const float* eps_ui0  = (const float*)d_in[13];
  const float* eps_iu0  = (const float*)d_in[14];
  const float* stuE     = (const float*)d_in[15];
  const float* exeE     = (const float*)d_in[16];
  const float* knE      = (const float*)d_in[17];
  const float* stu_b    = (const float*)d_in[18];
  const float* exer_b   = (const float*)d_in[19];
  const float* disc     = (const float*)d_in[20];
  const float* W1_l1    = (const float*)d_in[21];
  const float* b1_l1    = (const float*)d_in[22];
  const float* W2_l1    = (const float*)d_in[23];
  const float* b2_l1    = (const float*)d_in[24];
  const float* W1_l0    = (const float*)d_in[25];
  const float* b1_l0    = (const float*)d_in[26];
  const float* W2_l0    = (const float*)d_in[27];
  const float* b2_l0    = (const float*)d_in[28];
  const float* pw1      = (const float*)d_in[29];
  const float* pb1      = (const float*)d_in[30];
  const float* pw2      = (const float*)d_in[31];
  const float* pb2      = (const float*)d_in[32];
  const float* pw3      = (const float*)d_in[33];
  const float* pb3      = (const float*)d_in[34];

  char* ws = (char*)d_ws;
  size_t off = 0;
  auto alloc = [&](size_t bytes) -> void* {
    off = (off + 255) & ~(size_t)255;
    void* p = ws + off;
    off += bytes;
    return p;
  };

  _Float16* W1h1 = (_Float16*)alloc((size_t)D * 2 * D * 2);
  _Float16* W1h0 = (_Float16*)alloc((size_t)D * 2 * D * 2);
  _Float16* knh  = (_Float16*)alloc((size_t)128 * D * 2);
  _Float16* pw1h = (_Float16*)alloc((size_t)256 * 128 * 2);
  _Float16* pw2h = (_Float16*)alloc((size_t)128 * 256 * 2);
  float* v_ui1 = (float*)alloc((size_t)NE * 4);
  float* v_iu1 = (float*)alloc((size_t)NE * 4);
  float* v_ui0 = (float*)alloc((size_t)NE * 4);
  float* v_iu0 = (float*)alloc((size_t)NE * 4);
  float* stuA   = (float*)alloc((size_t)S * D * 4);
  float* stuB   = (float*)alloc((size_t)S * D * 4);
  float* stuSum = (float*)alloc((size_t)S * D * 4);
  float* exeA   = (float*)alloc((size_t)E * D * 4);
  float* exeB   = (float*)alloc((size_t)E * D * 4);
  float* exeSum = (float*)alloc((size_t)E * D * 4);
  _Float16* xh  = (_Float16*)alloc((size_t)B * 128 * 2);
  _Float16* h1h = (_Float16*)alloc((size_t)B * 256 * 2);
  _Float16* h2h = (_Float16*)alloc((size_t)B * 128 * 2);
  (void)ws_size; (void)in_sizes; (void)n_in; (void)out_size;

  // --- weight conversions to f16 (PosLinear |W| folded in) ---
  k_cvt_f16<<<(32768 + 255) / 256, 256, 0, stream>>>(W1h1, W1_l1, 32768, 0);
  k_cvt_f16<<<(32768 + 255) / 256, 256, 0, stream>>>(W1h0, W1_l0, 32768, 0);
  k_cvt_f16<<<(16384 + 255) / 256, 256, 0, stream>>>(knh,  knE,   16384, 0);
  k_cvt_f16<<<(32768 + 255) / 256, 256, 0, stream>>>(pw1h, pw1,   32768, 1);
  k_cvt_f16<<<(32768 + 255) / 256, 256, 0, stream>>>(pw2h, pw2,   32768, 1);

  // --- edge gates (shared logit per polarity, two eps streams) ---
  const int gateBlocks = NE / (16 * GW);   // 12500
  k_gate<<<gateBlocks, 128, 0, stream>>>(ui1_u, ui1_i, ui1_val, iu1_val, eps_ui1, eps_iu1,
                                         stuE, exeE, W1h1, b1_l1, W2_l1, b2_l1,
                                         v_ui1, v_iu1, NE);
  k_gate<<<gateBlocks, 128, 0, stream>>>(ui0_u, ui0_i, ui0_val, iu0_val, eps_ui0, eps_iu0,
                                         stuE, exeE, W1h0, b1_l0, W2_l0, b2_l0,
                                         v_ui0, v_iu0, NE);

  // --- GCN propagation: 3 layers, sum over layers ---
  const int nS = S * D, nEl = E * D;
  k_copy2<<<(nS + 255) / 256, 256, 0, stream>>>(stuA, stuSum, stuE, nS);
  k_copy2<<<(nEl + 255) / 256, 256, 0, stream>>>(exeA, exeSum, exeE, nEl);

  float *scur = stuA, *snxt = stuB, *ecur = exeA, *enxt = exeB;
  const int spmmBlocks = NE / 8;           // 8 waves (edges) per 256-thread block
  for (int l = 0; l < 3; ++l) {
    k_zero<<<(nS + 255) / 256, 256, 0, stream>>>(snxt, nS);
    k_zero<<<(nEl + 255) / 256, 256, 0, stream>>>(enxt, nEl);
    k_spmm<<<spmmBlocks, 256, 0, stream>>>(ui1_u, ui1_i, v_ui1, ecur, snxt, NE);
    k_spmm<<<spmmBlocks, 256, 0, stream>>>(ui0_u, ui0_i, v_ui0, ecur, snxt, NE);
    k_spmm<<<spmmBlocks, 256, 0, stream>>>(ui1_i, ui1_u, v_iu1, scur, enxt, NE);
    k_spmm<<<spmmBlocks, 256, 0, stream>>>(ui0_i, ui0_u, v_iu0, scur, enxt, NE);
    k_add<<<(nS + 255) / 256, 256, 0, stream>>>(stuSum, snxt, nS);
    k_add<<<(nEl + 255) / 256, 256, 0, stream>>>(exeSum, enxt, nEl);
    float* t;
    t = scur; scur = snxt; snxt = t;
    t = ecur; ecur = enxt; enxt = t;
  }

  // --- prediction head ---
  const int headBlocks = B / (16 * HW);    // 128
  k_head1<<<headBlocks, 128, 0, stream>>>(stu_id, exer_id, kn_emb, stuSum, exeSum, knh,
                                          stu_b, exer_b, disc, xh, B);
  k_gemm_sig<<<headBlocks, 128, 0, stream>>>(xh,  pw1h, pb1, h1h, 128, 256, B);
  k_gemm_sig<<<headBlocks, 128, 0, stream>>>(h1h, pw2h, pb2, h2h, 256, 128, B);
  k_out<<<(B + 255) / 256, 256, 0, stream>>>(h2h, pw3, pb3, (float*)d_out, B);
}
